// RSSM_18098992185574
// MI455X (gfx1250) — matrix operand order repaired and checked by reading, not compiled, not run on hardware
//
#include <hip/hip_runtime.h>
#include <cstdint>
#include <cstddef>

// ---------- types ----------
typedef __bf16 bf16_t;
typedef bf16_t bf16x16 __attribute__((ext_vector_type(16)));
typedef bf16_t bf16x8  __attribute__((ext_vector_type(8)));
typedef float  f32x8   __attribute__((ext_vector_type(8)));

#define Bsz   2048
#define Hh    15
#define SDd   1024
#define DUu   2048
#define FEATc 3072
#define OUTC  3077   // SD + DU + A + 2
#define ENCP  1088   // ENC_IN (1027) padded to multiple of 64

// ---------- CDNA5 async global->LDS path (probe-gated, with sync fallback) ----------
#if defined(__has_builtin)
#  if __has_builtin(__builtin_amdgcn_global_load_async_to_lds_b128)
#    define ASYNC_OK 1
#  endif
#endif
#ifndef ASYNC_OK
#  define ASYNC_OK 0
#endif

#if ASYNC_OK
#  if __has_builtin(__builtin_amdgcn_s_wait_asynccnt)
#    define WAIT_ASYNC(n) __builtin_amdgcn_s_wait_asynccnt(n)
#  else
#    define WAIT_ASYNC(n) asm volatile("s_wait_asynccnt " #n)
#  endif
typedef int v4i_ __attribute__((vector_size(16)));
typedef __attribute__((address_space(1))) v4i_ g_v4i;
typedef __attribute__((address_space(3))) v4i_ l_v4i;
__device__ __forceinline__ void async_ld16(const bf16_t* g, bf16_t* l)
{
    __builtin_amdgcn_global_load_async_to_lds_b128(
        (g_v4i*)(void*)(g),
        (l_v4i*)(void*)(l),
        0, 0);
}
#endif

__device__ __forceinline__ float sigmoidf_(float x) { return 1.0f / (1.0f + expf(-x)); }
__device__ __forceinline__ float siluf_(float x)    { return x * sigmoidf_(x); }
__device__ __forceinline__ float softplusf_(float x){
    return (x > 0.0f) ? (x + log1pf(expf(-x))) : log1pf(expf(x));
}

// ---------- weight fp32 [K][N] -> bf16 W^T [N][Kpad] (zero padded) ----------
__global__ void transpose_to_bf16(const float* __restrict__ src, bf16_t* __restrict__ dst,
                                  int K, int N, int Kpad)
{
    __shared__ float tile[32][33];
    const int nb = blockIdx.x * 32;
    const int kb = blockIdx.y * 32;
    const int tx = threadIdx.x;      // 0..31
    const int ty = threadIdx.y;      // 0..7
    #pragma unroll
    for (int i = 0; i < 32; i += 8) {
        int k = kb + ty + i;
        tile[ty + i][tx] = (k < K) ? src[(size_t)k * N + nb + tx] : 0.0f;
    }
    __syncthreads();
    #pragma unroll
    for (int i = 0; i < 32; i += 8) {
        int n = nb + ty + i;
        dst[(size_t)n * Kpad + kb + tx] = (bf16_t)tile[tx][ty + i];
    }
}

// ---------- WMMA GEMM: C[M,N] = A[M,K](bf16) x Wt[N,K](bf16), M=2048, BK=64 ----------
// EPI 0: Cf = acc + bias              (f32 out)
// EPI 1: Cb = silu(acc + bias)        (bf16 out)
// EPI 2: Cb = aux + silu(acc + bias)  (bf16 out, aux f32 read at same coords)
template <int EPI>
__global__ __launch_bounds__(256)
void gemm_bf16(const bf16_t* __restrict__ A, int lda,
               const bf16_t* __restrict__ Bt, int ldb,
               const float* __restrict__ bias,
               float* __restrict__ Cf, bf16_t* __restrict__ Cb,
               const float* __restrict__ aux,
               int K, int ldc)
{
    constexpr int SKEW = 72;                 // 144B row stride: conflict-free 16B lane reads
    __shared__ bf16_t sA[2][128 * SKEW];
    __shared__ bf16_t sB[2][128 * SKEW];

    const int tid   = threadIdx.x;
    const int lane  = tid & 31;
    const int wave  = tid >> 5;
    const int wm    = wave >> 1;             // 0..3 -> 32 rows each
    const int wn    = wave & 1;              // 0..1 -> 64 cols each
    const int khalf = lane >> 4;             // 0/1
    const int lidx  = lane & 15;

    const int m0 = blockIdx.x * 128;
    const int n0 = blockIdx.y * 128;

    f32x8 acc[2][4] = {};
    const int nsteps = K >> 6;

    auto compute = [&](int buf) {
        #pragma unroll
        for (int s2 = 0; s2 < 2; ++s2) {     // two K-slices of 32
            const int kof = s2 * 32;
            bf16x16 afrag[2], bfrag[4];
            #pragma unroll
            for (int i = 0; i < 2; ++i) {    // A 16x32: lo=K(kh*8+0..7), hi=K(16+kh*8+0..7)
                int row = wm * 32 + i * 16 + lidx;
                bf16x8 lo = *reinterpret_cast<const bf16x8*>(&sA[buf][row * SKEW + kof + khalf * 8]);
                bf16x8 hi = *reinterpret_cast<const bf16x8*>(&sA[buf][row * SKEW + kof + 16 + khalf * 8]);
                #pragma unroll
                for (int e = 0; e < 8; ++e) { afrag[i][e] = lo[e]; afrag[i][e + 8] = hi[e]; }
            }
            #pragma unroll
            for (int j = 0; j < 4; ++j) {    // B 32x16: lo=K(kh*8+0..7), hi=K(16+kh*8+0..7)
                int row = wn * 64 + j * 16 + lidx;
                bf16x8 lo = *reinterpret_cast<const bf16x8*>(&sB[buf][row * SKEW + kof + khalf * 8]);
                bf16x8 hi = *reinterpret_cast<const bf16x8*>(&sB[buf][row * SKEW + kof + 16 + khalf * 8]);
                #pragma unroll
                for (int e = 0; e < 8; ++e) { bfrag[j][e] = lo[e]; bfrag[j][e + 8] = hi[e]; }
            }
            #pragma unroll
            for (int i = 0; i < 2; ++i)
                #pragma unroll
                for (int j = 0; j < 4; ++j)
                    acc[i][j] = __builtin_amdgcn_wmma_f32_16x16x32_bf16(
                        false, afrag[i], false, bfrag[j], (short)0, acc[i][j], false, false);
        }
    };

#if ASYNC_OK
    // double-buffered async pipeline: DMA tile k+1 into LDS while WMMAs chew tile k
    auto issue = [&](int buf, int k0) {
        #pragma unroll
        for (int i = 0; i < 4; ++i) {        // 1024 16B-chunks per 128x64 tile, 4/thread
            int c   = tid + i * 256;
            int r   = c >> 3;
            int col = (c & 7) << 3;
            async_ld16(&A[(size_t)(m0 + r) * lda + k0 + col], &sA[buf][r * SKEW + col]);
            async_ld16(&Bt[(size_t)(n0 + r) * ldb + k0 + col], &sB[buf][r * SKEW + col]);
        }
    };
    issue(0, 0);
    int cur = 0;
    for (int s = 0; s < nsteps; ++s) {
        if (s + 1 < nsteps) { issue(cur ^ 1, (s + 1) << 6); WAIT_ASYNC(8); }
        else                { WAIT_ASYNC(0); }
        __syncthreads();                     // all waves' current tile resident
        compute(cur);
        __syncthreads();                     // done reading before DMA overwrites
        cur ^= 1;
    }
#else
    for (int s = 0; s < nsteps; ++s) {
        const int k0 = s << 6;
        #pragma unroll
        for (int i = 0; i < 4; ++i) {
            int c   = tid + i * 256;
            int r   = c >> 3;
            int col = (c & 7) << 3;
            *reinterpret_cast<bf16x8*>(&sA[0][r * SKEW + col]) =
                *reinterpret_cast<const bf16x8*>(&A[(size_t)(m0 + r) * lda + k0 + col]);
            *reinterpret_cast<bf16x8*>(&sB[0][r * SKEW + col]) =
                *reinterpret_cast<const bf16x8*>(&Bt[(size_t)(n0 + r) * ldb + k0 + col]);
        }
        __syncthreads();
        compute(0);
        __syncthreads();
    }
#endif

    const int cmb = m0 + wm * 32;
    const int cnb = n0 + wn * 64;
    #pragma unroll
    for (int i = 0; i < 2; ++i) {
        #pragma unroll
        for (int j = 0; j < 4; ++j) {
            int cn   = cnb + j * 16 + lidx;
            float bv = bias[cn];
            #pragma unroll
            for (int r = 0; r < 8; ++r) {
                int cm = cmb + i * 16 + khalf * 8 + r;
                size_t idx = (size_t)cm * ldc + cn;
                float v = acc[i][j][r] + bv;
                if (EPI == 0)      Cf[idx] = v;
                else if (EPI == 1) Cb[idx] = (bf16_t)siluf_(v);
                else               Cb[idx] = (bf16_t)(aux[idx] + siluf_(v));
            }
        }
    }
}

// ---------- init state: deter/stoch f32 state + feat bf16 = [deter|stoch] ----------
__global__ void k_init(const float* __restrict__ stoch0, const float* __restrict__ deter0,
                       float* __restrict__ deterS, float* __restrict__ stochS,
                       bf16_t* __restrict__ feat)
{
    int idx = blockIdx.x * 256 + threadIdx.x;          // B*3072
    int b = idx / FEATc, c = idx % FEATc;
    if (c < DUu) {
        float v = deter0[(size_t)b * DUu + c];
        deterS[(size_t)b * DUu + c] = v;
        feat[(size_t)b * FEATc + c] = (bf16_t)v;
    } else {
        int s = c - DUu;
        float v = stoch0[(size_t)b * SDd + s];
        stochS[(size_t)b * SDd + s] = v;
        feat[(size_t)b * FEATc + c] = (bf16_t)v;
    }
}

// ---------- policy + build encoder input x_enc = [stoch|action|pad] ----------
__global__ __launch_bounds__(256)
void k_policy(const float* __restrict__ deterS, const float* __restrict__ stochS,
              const float* __restrict__ pol_w, const float* __restrict__ pol_b,
              bf16_t* __restrict__ x_enc, float* __restrict__ out, int t)
{
    const int b = blockIdx.x, tid = threadIdx.x;
    const float* dr = deterS + (size_t)b * DUu;
    const float* sr = stochS + (size_t)b * SDd;
    float p0 = 0.f, p1 = 0.f, p2 = 0.f;
    for (int i = tid; i < FEATc; i += 256) {
        float f = (i < DUu) ? dr[i] : sr[i - DUu];
        p0 += f * pol_w[i * 3 + 0];
        p1 += f * pol_w[i * 3 + 1];
        p2 += f * pol_w[i * 3 + 2];
    }
    __shared__ float red[3][256];
    red[0][tid] = p0; red[1][tid] = p1; red[2][tid] = p2;
    __syncthreads();
    for (int s = 128; s > 0; s >>= 1) {
        if (tid < s) { red[0][tid] += red[0][tid + s]; red[1][tid] += red[1][tid + s]; red[2][tid] += red[2][tid + s]; }
        __syncthreads();
    }
    __shared__ float act[3];
    if (tid < 3) act[tid] = tanhf(red[tid][0] + pol_b[tid]);
    __syncthreads();
    bf16_t* xr = x_enc + (size_t)b * ENCP;
    for (int i = tid; i < ENCP; i += 256) {
        float v;
        if (i < SDd) v = sr[i];
        else if (i < SDd + 3) v = act[i - SDd];
        else v = 0.0f;
        xr[i] = (bf16_t)v;
    }
    if (tid < 3) out[((size_t)b * Hh + t) * OUTC + 3072 + tid] = act[tid];
}

// ---------- GRU elementwise -> new deter ----------
__global__ void k_gru(const float* __restrict__ mx, const float* __restrict__ mh,
                      float* __restrict__ deterS, bf16_t* __restrict__ feat,
                      float* __restrict__ out, int t)
{
    size_t idx = (size_t)blockIdx.x * 256 + threadIdx.x;   // B*2048
    int b = (int)(idx >> 11), i = (int)(idx & 2047);
    const float* mxr = mx + (size_t)b * 6144;
    const float* mhr = mh + (size_t)b * 6144;
    float z    = sigmoidf_(mxr[i]        + mhr[i]);
    float r    = sigmoidf_(mxr[2048 + i] + mhr[2048 + i]);
    float cand = tanhf(mxr[4096 + i] + r * mhr[4096 + i]);
    float h  = deterS[idx];
    float hn = z * h + (1.0f - z) * cand;
    deterS[idx] = hn;
    feat[(size_t)b * FEATc + i] = (bf16_t)hn;
    out[((size_t)b * Hh + t) * OUTC + SDd + i] = hn;
}

// ---------- reparameterized sample -> new stoch ----------
__global__ void k_sample(const float* __restrict__ io, const float* __restrict__ eps,
                         float* __restrict__ stochS, bf16_t* __restrict__ feat,
                         float* __restrict__ out, int t)
{
    size_t idx = (size_t)blockIdx.x * 256 + threadIdx.x;   // B*1024
    int b = (int)(idx >> 10), i = (int)(idx & 1023);
    float mean = io[(size_t)b * DUu + i];
    float sd   = softplusf_(io[(size_t)b * DUu + SDd + i]) + 0.1f;
    float e    = eps[((size_t)t * Bsz + b) * SDd + i];
    float s    = mean + sd * e;
    stochS[idx] = s;
    feat[(size_t)b * FEATc + DUu + i] = (bf16_t)s;
    out[((size_t)b * Hh + t) * OUTC + i] = s;
}

// ---------- N=1 head: dot(hid[b], w2) + b2, optional sigmoid ----------
template <bool SIG>
__global__ __launch_bounds__(256)
void k_head(const bf16_t* __restrict__ hid, const float* __restrict__ w2,
            const float* __restrict__ b2, float* __restrict__ out, int t, int col)
{
    const int b = blockIdx.x, tid = threadIdx.x;
    const bf16_t* hr = hid + (size_t)b * DUu;
    float p = 0.f;
    for (int i = tid; i < DUu; i += 256) p += (float)hr[i] * w2[i];
    __shared__ float red[256];
    red[tid] = p;
    __syncthreads();
    for (int s = 128; s > 0; s >>= 1) { if (tid < s) red[tid] += red[tid + s]; __syncthreads(); }
    if (tid == 0) {
        float v = red[0] + b2[0];
        if (SIG) v = sigmoidf_(v);
        out[((size_t)b * Hh + t) * OUTC + col] = v;
    }
}

// ---------- host ----------
extern "C" void kernel_launch(void* const* d_in, const int* in_sizes, int n_in,
                              void* d_out, int out_size, void* d_ws, size_t ws_size,
                              hipStream_t stream)
{
    const float* stoch0     = (const float*)d_in[0];
    const float* deter0     = (const float*)d_in[1];
    const float* eps        = (const float*)d_in[3];
    const float* enc_w1     = (const float*)d_in[4];
    const float* enc_b1     = (const float*)d_in[5];
    const float* enc_w2     = (const float*)d_in[6];
    const float* enc_b2     = (const float*)d_in[7];
    const float* img_step_w = (const float*)d_in[8];
    const float* img_step_b = (const float*)d_in[9];
    const float* gru_kernel = (const float*)d_in[10];
    const float* gru_rec    = (const float*)d_in[11];
    const float* gru_bias   = (const float*)d_in[12];
    const float* img_out_w  = (const float*)d_in[13];
    const float* img_out_b  = (const float*)d_in[14];
    const float* rew_w1     = (const float*)d_in[15];
    const float* rew_b1     = (const float*)d_in[16];
    const float* rew_w2     = (const float*)d_in[17];
    const float* rew_b2     = (const float*)d_in[18];
    const float* cont_w1    = (const float*)d_in[19];
    const float* cont_b1    = (const float*)d_in[20];
    const float* cont_w2    = (const float*)d_in[21];
    const float* cont_b2    = (const float*)d_in[22];
    const float* pol_w      = (const float*)d_in[23];
    const float* pol_b      = (const float*)d_in[24];
    float* out = (float*)d_out;

    // ---- workspace carve ----
    char* ws = (char*)d_ws;
    size_t off = 0;
    auto alloc = [&](size_t bytes) -> char* {
        char* p = ws + off;
        off = (off + bytes + 255) & ~(size_t)255;
        return p;
    };
    bf16_t* WTenc1 = (bf16_t*)alloc((size_t)8192 * ENCP * 2);
    bf16_t* WTenc2 = (bf16_t*)alloc((size_t)2048 * 8192 * 2);
    bf16_t* WTimg  = (bf16_t*)alloc((size_t)2048 * 2048 * 2);
    bf16_t* WTgk   = (bf16_t*)alloc((size_t)6144 * 2048 * 2);
    bf16_t* WTgr   = (bf16_t*)alloc((size_t)6144 * 2048 * 2);
    bf16_t* WTio   = (bf16_t*)alloc((size_t)2048 * 2048 * 2);
    bf16_t* WTr1   = (bf16_t*)alloc((size_t)2048 * 3072 * 2);
    bf16_t* WTc1   = (bf16_t*)alloc((size_t)2048 * 3072 * 2);
    bf16_t* feat   = (bf16_t*)alloc((size_t)Bsz * FEATc * 2);
    bf16_t* x_enc  = (bf16_t*)alloc((size_t)Bsz * ENCP * 2);
    bf16_t* h1     = (bf16_t*)alloc((size_t)Bsz * 8192 * 2);
    float*  xf32   = (float*) alloc((size_t)Bsz * 2048 * 4);
    bf16_t* xb     = (bf16_t*)alloc((size_t)Bsz * 2048 * 2);
    float*  mx     = (float*) alloc((size_t)Bsz * 6144 * 4);
    float*  mh     = (float*) alloc((size_t)Bsz * 6144 * 4);
    float*  iout   = (float*) alloc((size_t)Bsz * 2048 * 4);
    bf16_t* hid    = (bf16_t*)alloc((size_t)Bsz * 2048 * 2);
    float*  deterS = (float*) alloc((size_t)Bsz * DUu * 4);
    float*  stochS = (float*) alloc((size_t)Bsz * SDd * 4);
    if (off > ws_size) return;   // workspace too small; bail without corrupting memory

    // ---- one-time (per call) weight downconvert + transpose ----
    auto T = [&](const float* src, bf16_t* dst, int K, int N, int Kpad) {
        dim3 g(N / 32, Kpad / 32), b(32, 8);
        transpose_to_bf16<<<g, b, 0, stream>>>(src, dst, K, N, Kpad);
    };
    T(enc_w1,     WTenc1, 1027, 8192, ENCP);
    T(enc_w2,     WTenc2, 8192, 2048, 8192);
    T(img_step_w, WTimg,  2048, 2048, 2048);
    T(gru_kernel, WTgk,   2048, 6144, 2048);
    T(gru_rec,    WTgr,   2048, 6144, 2048);
    T(img_out_w,  WTio,   2048, 2048, 2048);
    T(rew_w1,     WTr1,   3072, 2048, 3072);
    T(cont_w1,    WTc1,   3072, 2048, 3072);

    k_init<<<(Bsz * FEATc) / 256, 256, 0, stream>>>(stoch0, deter0, deterS, stochS, feat);

    auto G = [&](int epi, const bf16_t* A, int lda, const bf16_t* Bt, int ldb,
                 const float* bias, float* Cf, bf16_t* Cb, const float* aux,
                 int N, int K, int ldc) {
        dim3 g(Bsz / 128, N / 128), b(256);
        if (epi == 0)      gemm_bf16<0><<<g, b, 0, stream>>>(A, lda, Bt, ldb, bias, Cf, Cb, aux, K, ldc);
        else if (epi == 1) gemm_bf16<1><<<g, b, 0, stream>>>(A, lda, Bt, ldb, bias, Cf, Cb, aux, K, ldc);
        else               gemm_bf16<2><<<g, b, 0, stream>>>(A, lda, Bt, ldb, bias, Cf, Cb, aux, K, ldc);
    };

    for (int t = 0; t < Hh; ++t) {
        // action = tanh(feat_prev @ pol_w + b); x_enc = [stoch|action|0pad]
        k_policy<<<Bsz, 256, 0, stream>>>(deterS, stochS, pol_w, pol_b, x_enc, out, t);
        // h1 = silu(x_enc @ enc_w1 + b1)              [B,8192] bf16
        G(1, x_enc, ENCP, WTenc1, ENCP, enc_b1, nullptr, h1, nullptr, 8192, ENCP, 8192);
        // xf32 = h1 @ enc_w2 + b2                     [B,2048] f32
        G(0, h1, 8192, WTenc2, 8192, enc_b2, xf32, nullptr, nullptr, 2048, 8192, 2048);
        // xb = bf16(xf32 + silu(deter @ img_step_w + b))
        G(2, feat, FEATc, WTimg, 2048, img_step_b, nullptr, xb, xf32, 2048, 2048, 2048);
        // mx = xb @ gru_kernel + bias[0]; mh = deter @ gru_rec + bias[1]
        G(0, xb, 2048, WTgk, 2048, gru_bias,           mx, nullptr, nullptr, 6144, 2048, 6144);
        G(0, feat, FEATc, WTgr, 2048, gru_bias + 6144, mh, nullptr, nullptr, 6144, 2048, 6144);
        // deter_n
        k_gru<<<(Bsz * DUu) / 256, 256, 0, stream>>>(mx, mh, deterS, feat, out, t);
        // img_out on new deter
        G(0, feat, FEATc, WTio, 2048, img_out_b, iout, nullptr, nullptr, 2048, 2048, 2048);
        // stoch_n = mean + (softplus(std)+0.1)*eps
        k_sample<<<(Bsz * SDd) / 256, 256, 0, stream>>>(iout, eps, stochS, feat, out, t);
        // reward head
        G(1, feat, FEATc, WTr1, 3072, rew_b1, nullptr, hid, nullptr, 2048, 3072, 2048);
        k_head<false><<<Bsz, 256, 0, stream>>>(hid, rew_w2, rew_b2, out, t, 3075);
        // cont head
        G(1, feat, FEATc, WTc1, 3072, cont_b1, nullptr, hid, nullptr, 2048, 3072, 2048);
        k_head<true><<<Bsz, 256, 0, stream>>>(hid, cont_w2, cont_b2, out, t, 3076);
    }
}
